// MultiHeadAttention_9371618640081
// MI455X (gfx1250) — compile-verified, matrix operand order checked by reading, not run on hardware
//
#include <hip/hip_runtime.h>
#include <hip/hip_bf16.h>

// ---------------- problem constants ----------------
#define Bb   2
#define Tt   2048
#define Cc   1024
#define Hh   16
#define HS   64
#define NH_HS (Hh*HS)          // 1024
#define SCALE 0.125f           // 1/sqrt(64)

// ---------------- vector types ----------------
typedef __attribute__((ext_vector_type(16))) __bf16 v16bf;
typedef __attribute__((ext_vector_type(8)))  __bf16 bf16x8;
typedef __attribute__((ext_vector_type(8)))  float  v8f;

union FragU { v16bf v; bf16x8 h[2]; };
union Pack8 { bf16x8 v; __bf16 e[8]; };

// Load a 16-element bf16 WMMA fragment whose per-lane elements are
// row-contiguous in memory (A-fragment of a row-major matrix, or
// B-fragment of a K-contiguous transposed matrix).
// Element e -> k = (e<8 ? half*8+e : 16+half*8+(e-8))  (ISA 16-bit A layout)
__device__ inline v16bf load_frag_row(const __bf16* row, int kbase, int half) {
  FragU f;
  f.h[0] = *(const bf16x8*)(row + kbase + half * 8);
  f.h[1] = *(const bf16x8*)(row + kbase + 16 + half * 8);
  return f.v;
}

__device__ inline v8f wmma_bf16(v16bf a, v16bf b, v8f c) {
  // D = A*B + C, fp32 accumulate
  return __builtin_amdgcn_wmma_f32_16x16x32_bf16(
      false, a, false, b, (short)0, c, false, false);
}

// ---------------- conversion kernels ----------------
__global__ void cvt_f32_bf16_kernel(const float* __restrict__ src,
                                    __bf16* __restrict__ dst, int n) {
  int i = blockIdx.x * blockDim.x + threadIdx.x;
  if (i < n) dst[i] = (__bf16)src[i];
}

// src: [Hn][Rows][Cols] f32 -> dst: [Hn][Cols][Rows] bf16 (per-slice transpose)
__global__ void cvt_transpose_bf16_kernel(const float* __restrict__ src,
                                          __bf16* __restrict__ dst,
                                          int Hn, int Rows, int Cols) {
  long i = (long)blockIdx.x * blockDim.x + threadIdx.x;
  long total = (long)Hn * Rows * Cols;
  if (i >= total) return;
  int  r   = (int)(i % Rows);          // fastest -> coalesced dst writes
  long tmp = i / Rows;
  int  c   = (int)(tmp % Cols);
  int  h   = (int)(tmp / Cols);
  dst[((long)h * Cols + c) * Rows + r] = (__bf16)src[((long)h * Rows + r) * Cols + c];
}

// ---------------- QKV projection (split: blockIdx.y selects Q/K/V) ----------
// xb:[B*T, C] bf16 ; Wt:[H, HS, C] bf16 (pre-transposed)
// Q,K:[B*H, T, HS] bf16 ; VT:[B*H, HS, T] bf16
// One wave computes a 16(M) x 64(N) tile of ONE of Q/K/V for one (b,h).
// 4 accumulators (32 VGPRs) -> no scratch spills; x is 8 MB bf16, so the
// 3x re-read stays in the 192 MB L2.
__global__ void __launch_bounds__(256)
qkv_proj_kernel(const __bf16* __restrict__ xb,
                const __bf16* __restrict__ Wqt,
                const __bf16* __restrict__ Wkt,
                const __bf16* __restrict__ Wvt,
                __bf16* __restrict__ Q,
                __bf16* __restrict__ K,
                __bf16* __restrict__ VT) {
  const int lane = threadIdx.x & 31;
  const int idx  = lane & 15;
  const int half = lane >> 4;
  const int wg   = blockIdx.x * (blockDim.x >> 5) + (threadIdx.x >> 5);
  const int mtile = wg & 127;        // T/16 = 128
  const int bh    = wg >> 7;         // 0..31
  const int b     = bh >> 4;
  const int which = blockIdx.y;      // 0=Q, 1=K, 2=V

  const __bf16* Wt = (which == 0) ? Wqt : ((which == 1) ? Wkt : Wvt);
  const __bf16* xrow = xb + ((size_t)(b * Tt) + mtile * 16 + idx) * Cc;
  const __bf16* wb   = Wt + ((size_t)(bh & 15) * HS) * Cc;

  v8f acc[4] = {};
#pragma unroll 2
  for (int kk = 0; kk < Cc; kk += 32) {
    v16bf a = load_frag_row(xrow, kk, half);
#pragma unroll
    for (int nt = 0; nt < 4; ++nt) {
      v16bf bf = load_frag_row(wb + (size_t)(nt * 16 + idx) * Cc, kk, half);
      acc[nt] = wmma_bf16(a, bf, acc[nt]);
    }
  }

  if (which == 2) {
    // store V transposed [bh][d][t]; 8 consecutive t -> one 16B store
#pragma unroll
    for (int nt = 0; nt < 4; ++nt) {
      Pack8 pk;
#pragma unroll
      for (int r = 0; r < 8; ++r) pk.e[r] = (__bf16)acc[nt][r];
      *(bf16x8*)(VT + ((size_t)bh * HS + nt * 16 + idx) * Tt +
                 mtile * 16 + 8 * half) = pk.v;
    }
  } else {
    __bf16* Out = (which == 0) ? Q : K;   // row-major [bh][t][d]
#pragma unroll
    for (int nt = 0; nt < 4; ++nt) {
#pragma unroll
      for (int r = 0; r < 8; ++r) {
        int t = mtile * 16 + r + 8 * half;
        Out[((size_t)bh * Tt + t) * HS + nt * 16 + idx] = (__bf16)acc[nt][r];
      }
    }
  }
}

// ---------------- flash attention (block-cooperative K/V staging) ----------
// Block = 4 waves = 4 consecutive 16-row query tiles of one (b,h).
// Each 32-wide key block is staged into LDS ONCE per block (4 KB K + 4 KB V),
// cutting L2->WGP K/V traffic 4x; all B-fragments then come from LDS
// (ds_load_b128, conflict-free). Online softmax keeps fully-masked blocks
// exact no-ops, so the block-uniform trip count is safe for __syncthreads.
__global__ void __launch_bounds__(128)
attn_kernel(const __bf16* __restrict__ Q,
            const __bf16* __restrict__ K,
            const __bf16* __restrict__ VT,
            __bf16* __restrict__ Y) {
  __shared__ __align__(16) __bf16 ldsK[32 * 64];      // [s_local][d]
  __shared__ __align__(16) __bf16 ldsV[64 * 32];      // [d][s_local]
  __shared__ __align__(16) __bf16 ldsP[4][16 * 32];   // per-wave P staging
  const int tid  = threadIdx.x;
  const int lane = tid & 31;
  const int idx  = lane & 15;
  const int half = lane >> 4;
  const int warp = tid >> 5;
  const int bh     = blockIdx.x >> 5;       // 32 q-groups per (b,h)
  const int qgroup = blockIdx.x & 31;
  const int tq     = qgroup * 4 + warp;     // this wave's query tile
  const int b      = bh >> 4;
  const int h      = bh & 15;

  // preload Q A-fragments (hs = 64 -> 2 fragments)
  const __bf16* qrow = Q + ((size_t)bh * Tt + tq * 16 + idx) * HS;
  v16bf qa0 = load_frag_row(qrow, 0, half);
  v16bf qa1 = load_frag_row(qrow, 32, half);

  v8f   oacc[4] = {};
  float mrow[8], lrow[8];
#pragma unroll
  for (int r = 0; r < 8; ++r) { mrow[r] = -1e30f; lrow[r] = 0.f; }

  const int smax = qgroup * 64 + 64;        // block-uniform causal bound
  __bf16* lp = ldsP[warp];

  for (int s0 = 0; s0 < smax; s0 += 32) {
    // ---- cooperative staging: 128 threads move 8 KB (K: 256 chunks,
    //      V: 256 chunks of 8 bf16 = 16 B each; 2+2 chunks per thread) ----
#pragma unroll
    for (int j = 0; j < 2; ++j) {
      int c = tid * 2 + j;                  // 0..255
      int kr = c >> 3, ko = (c & 7) * 8;    // K: 32 rows x 64
      *(bf16x8*)(ldsK + kr * 64 + ko) =
          *(const bf16x8*)(K + ((size_t)bh * Tt + s0 + kr) * HS + ko);
      int vd = c >> 2, vo = (c & 3) * 8;    // V: 64 rows x 32
      *(bf16x8*)(ldsV + vd * 32 + vo) =
          *(const bf16x8*)(VT + ((size_t)bh * HS + vd) * Tt + s0 + vo);
    }
    __builtin_prefetch(K + ((size_t)bh * Tt + s0 + 32 + (tid >> 3)) * HS, 0, 1);
    __syncthreads();

    // ---- S = Q K^T for two 16-col tiles (B-fragments from LDS) ----
    v8f sacc0 = {}, sacc1 = {};
    {
      v16bf b0 = load_frag_row(ldsK + idx * 64, 0, half);
      sacc0 = wmma_bf16(qa0, b0, sacc0);
      v16bf b1 = load_frag_row(ldsK + (16 + idx) * 64, 0, half);
      sacc1 = wmma_bf16(qa0, b1, sacc1);
      b0 = load_frag_row(ldsK + idx * 64, 32, half);
      sacc0 = wmma_bf16(qa1, b0, sacc0);
      b1 = load_frag_row(ldsK + (16 + idx) * 64, 32, half);
      sacc1 = wmma_bf16(qa1, b1, sacc1);
    }

    // ---- scale + causal mask + online softmax ----
    float p0v[8], p1v[8];
#pragma unroll
    for (int r = 0; r < 8; ++r) {
      int   tg = tq * 16 + r + 8 * half;
      float a0 = sacc0[r] * SCALE;
      float a1 = sacc1[r] * SCALE;
      float s0v = (s0 + idx      <= tg) ? a0 : -1e30f;
      float s1v = (s0 + 16 + idx <= tg) ? a1 : -1e30f;
      float rm = fmaxf(s0v, s1v);
      rm = fmaxf(rm, __shfl_xor(rm, 1));
      rm = fmaxf(rm, __shfl_xor(rm, 2));
      rm = fmaxf(rm, __shfl_xor(rm, 4));
      rm = fmaxf(rm, __shfl_xor(rm, 8));
      float mnew = fmaxf(mrow[r], rm);
      float cf   = __expf(mrow[r] - mnew);
      mrow[r] = mnew;
      float p0 = __expf(s0v - mnew);
      float p1 = __expf(s1v - mnew);
      float rs = p0 + p1;
      rs += __shfl_xor(rs, 1);
      rs += __shfl_xor(rs, 2);
      rs += __shfl_xor(rs, 4);
      rs += __shfl_xor(rs, 8);
      lrow[r] = lrow[r] * cf + rs;
#pragma unroll
      for (int dt = 0; dt < 4; ++dt) oacc[dt][r] *= cf;
      p0v[r] = p0; p1v[r] = p1;
    }

    // ---- stage P (D layout) in per-wave LDS, reread as A fragment ----
#pragma unroll
    for (int r = 0; r < 8; ++r) {
      int m = r + 8 * half;
      lp[m * 32 + idx]      = (__bf16)p0v[r];
      lp[m * 32 + 16 + idx] = (__bf16)p1v[r];
    }
    __builtin_amdgcn_wave_barrier();        // DS in-order per wave
    v16bf pa = load_frag_row(lp + idx * 32, 0, half);
    __builtin_amdgcn_wave_barrier();

    // ---- O += P * V (B-fragments from LDS V tile) ----
#pragma unroll
    for (int dt = 0; dt < 4; ++dt) {
      v16bf bv = load_frag_row(ldsV + (dt * 16 + idx) * 32, 0, half);
      oacc[dt] = wmma_bf16(pa, bv, oacc[dt]);
    }
    __syncthreads();                        // before next staging round
  }

  // normalize + store into concat layout Y:[B, T, H*HS] bf16
#pragma unroll
  for (int dt = 0; dt < 4; ++dt) {
#pragma unroll
    for (int r = 0; r < 8; ++r) {
      int t = tq * 16 + r + 8 * half;
      float o = oacc[dt][r] / lrow[r];
      Y[((size_t)b * Tt + t) * NH_HS + h * HS + dt * 16 + idx] = (__bf16)o;
    }
  }
}

// ---------------- output projection ----------------
// out[B*T,1024] f32 = Y[B*T,1024]bf16 @ Wo + bo ; WoT:[N=1024][K=1024] bf16
__global__ void __launch_bounds__(256)
out_proj_kernel(const __bf16* __restrict__ Y,
                const __bf16* __restrict__ WoT,
                const float* __restrict__ bo,
                float* __restrict__ out) {
  const int lane = threadIdx.x & 31;
  const int idx  = lane & 15;
  const int half = lane >> 4;
  const int wg   = blockIdx.x * (blockDim.x >> 5) + (threadIdx.x >> 5);
  const int mtile = wg & 255;        // (B*T)/16 = 256
  const int nt64  = wg >> 8;         // 1024/64  = 16

  const __bf16* yrow = Y + ((size_t)(mtile * 16 + idx)) * NH_HS;
  v8f acc[4] = {};
#pragma unroll 2
  for (int kk = 0; kk < NH_HS; kk += 32) {
    v16bf a = load_frag_row(yrow, kk, half);
#pragma unroll
    for (int nt = 0; nt < 4; ++nt) {
      const __bf16* wrow = WoT + (size_t)(nt64 * 64 + nt * 16 + idx) * NH_HS;
      v16bf bf = load_frag_row(wrow, kk, half);
      acc[nt] = wmma_bf16(a, bf, acc[nt]);
    }
  }
#pragma unroll
  for (int nt = 0; nt < 4; ++nt) {
    int col = nt64 * 64 + nt * 16 + idx;
    float bias = bo[col];
#pragma unroll
    for (int r = 0; r < 8; ++r) {
      int row = mtile * 16 + r + 8 * half;
      out[(size_t)row * Cc + col] = acc[nt][r] + bias;
    }
  }
}

// ---------------- launch ----------------
extern "C" void kernel_launch(void* const* d_in, const int* in_sizes, int n_in,
                              void* d_out, int out_size, void* d_ws, size_t ws_size,
                              hipStream_t stream) {
  const float* x  = (const float*)d_in[0];
  const float* Wq = (const float*)d_in[1];
  const float* Wk = (const float*)d_in[2];
  const float* Wv = (const float*)d_in[3];
  const float* Wo = (const float*)d_in[4];
  const float* bo = (const float*)d_in[5];
  float* out = (float*)d_out;

  // workspace carve-up (all bf16 buffers, sizes in bytes)
  char* ws = (char*)d_ws;
  const size_t SZ_XB = (size_t)Bb * Tt * Cc * 2;           // 8 MB
  const size_t SZ_W  = (size_t)Hh * HS * Cc * 2;           // 2 MB each
  const size_t SZ_WO = (size_t)NH_HS * Cc * 2;             // 2 MB
  const size_t SZ_Q  = (size_t)Bb * Hh * Tt * HS * 2;      // 8 MB each
  size_t off = 0;
  __bf16* xb  = (__bf16*)(ws + off); off += SZ_XB;
  __bf16* Wqt = (__bf16*)(ws + off); off += SZ_W;
  __bf16* Wkt = (__bf16*)(ws + off); off += SZ_W;
  __bf16* Wvt = (__bf16*)(ws + off); off += SZ_W;
  __bf16* WoT = (__bf16*)(ws + off); off += SZ_WO;
  __bf16* Q   = (__bf16*)(ws + off); off += SZ_Q;
  __bf16* K   = (__bf16*)(ws + off); off += SZ_Q;
  __bf16* VT  = (__bf16*)(ws + off); off += SZ_Q;
  __bf16* Y   = (__bf16*)(ws + off); off += SZ_XB;

  // 1) precision conversion + weight transposes (one-time cost)
  {
    int n = Bb * Tt * Cc;
    cvt_f32_bf16_kernel<<<(n + 255) / 256, 256, 0, stream>>>(x, xb, n);
    int nw = Hh * Cc * HS;
    cvt_transpose_bf16_kernel<<<(nw + 255) / 256, 256, 0, stream>>>(Wq, Wqt, Hh, Cc, HS);
    cvt_transpose_bf16_kernel<<<(nw + 255) / 256, 256, 0, stream>>>(Wk, Wkt, Hh, Cc, HS);
    cvt_transpose_bf16_kernel<<<(nw + 255) / 256, 256, 0, stream>>>(Wv, Wvt, Hh, Cc, HS);
    int no = NH_HS * Cc;
    cvt_transpose_bf16_kernel<<<(no + 255) / 256, 256, 0, stream>>>(Wo, WoT, 1, NH_HS, Cc);
  }
  // 2) QKV projections (WMMA bf16, spill-free 4-accumulator waves;
  //    blockIdx.y picks Q / K / V-transposed)
  qkv_proj_kernel<<<dim3(512, 3), 256, 0, stream>>>(xb, Wqt, Wkt, Wvt, Q, K, VT);
  // 3) flash attention (WMMA QK^T + online softmax + WMMA PV,
  //    block-cooperative LDS staging of K/V)
  attn_kernel<<<1024, 128, 0, stream>>>(Q, K, VT, Y);
  // 4) output projection + bias (fp32 out)
  out_proj_kernel<<<512, 256, 0, stream>>>(Y, WoT, bo, out);
}
